// MultiHeadAttention_87814901334366
// MI455X (gfx1250) — matrix operand order repaired and checked by reading, not compiled, not run on hardware
//
#include <hip/hip_runtime.h>
#include <hip/hip_bf16.h>

// ---------------------------------------------------------------------------
// MHA forward for MI455X (gfx1250, wave32, WMMA f16 -> f32 accumulate)
// B=4, S=2048, D=1024, H=16, HD=64.  Intermediates held in f16; K/V attention
// tiles staged into padded LDS via the Tensor Data Mover (TDM).
// ---------------------------------------------------------------------------

typedef _Float16 half8  __attribute__((ext_vector_type(8)));
typedef _Float16 v16h   __attribute__((ext_vector_type(16)));
typedef float    v8f    __attribute__((ext_vector_type(8)));
typedef unsigned int u32x4 __attribute__((ext_vector_type(4)));
typedef int      i32x4 __attribute__((ext_vector_type(4)));
typedef int      i32x8 __attribute__((ext_vector_type(8)));

#define CAT16(lo, hi) __builtin_shufflevector((lo), (hi), 0,1,2,3,4,5,6,7,8,9,10,11,12,13,14,15)
#define WMMA_F16(a, b, c) \
    __builtin_amdgcn_wmma_f32_16x16x32_f16(false, (a), false, (b), (short)0, (c), false, false)

static constexpr int B_ = 4;
static constexpr int S_ = 2048;
static constexpr int D_ = 1024;
static constexpr int H_ = 16;
static constexpr int GM = B_ * S_;   // 8192
static constexpr int GK = D_;        // 1024
static constexpr int GN = D_;        // 1024

// ---------------------------------------------------------------------------
// TDM: 2-D f16 tile (width x rows, row stride in elements) -> padded LDS.
// LDS layout gets 16B of pad per 128B row (rows of 64 f16 -> stride 72 f16),
// expressed as pad_interval = 32 DWORDs (code 4), pad_amount = 4 DWORDs (code 3).
// D# built per CDNA5 ISA 8.3/8.4 (group0: count/lds/global/type=2; group1:
// data_size=2B, tensor dims, tile dims, dim0 stride, pad config).
// ---------------------------------------------------------------------------
__device__ __forceinline__ void tdm_load_tile_f16(
    unsigned int lds_off, const void* gptr,
    unsigned int width, unsigned int rows, unsigned int stride_elems)
{
    const unsigned long long ga = (unsigned long long)(uintptr_t)gptr;
    u32x4 g0;
    g0.x = 1u;                                    // count=1, user mode
    g0.y = lds_off;                               // lds_addr (bytes)
    g0.z = (unsigned int)ga;                      // global_addr[31:0]
    g0.w = (unsigned int)(ga >> 32) | (2u << 30); // global_addr[56:32] | type=2
    i32x8 g1;
    g1[0] = (1 << 16) | (1 << 20) | (4 << 22) | (3 << 25); // 2B, pad_en, 32dw/4dw
    g1[1] = (int)(width << 16);                   // tensor_dim0[15:0]
    g1[2] = (int)(rows << 16);                    // dim0 hi=0 | tensor_dim1[15:0]
    g1[3] = (int)(width << 16);                   // dim1 hi=0 | tile_dim0
    g1[4] = (int)(rows & 0xFFFF);                 // tile_dim1 | tile_dim2=0
    g1[5] = (int)stride_elems;                    // tensor_dim0_stride[31:0]
    g1[6] = 0;
    g1[7] = 0;
    const i32x4 z4 = {0, 0, 0, 0};
#if defined(__clang_major__) && (__clang_major__ >= 23)
    const i32x8 z8 = {0, 0, 0, 0, 0, 0, 0, 0};
    __builtin_amdgcn_tensor_load_to_lds(g0, g1, z4, z4, z8, 0);
#else
    __builtin_amdgcn_tensor_load_to_lds(g0, g1, z4, z4, 0);
#endif
}

// ---------------------------------------------------------------------------
// GEMM: C[M,N] = A[M,K] @ W[K,N] + bias, f16 WMMA math, f32 accumulate.
// block = 256 threads = 8 waves (4 in M x 2 in N), block tile 128x128,
// wave tile 32x64 (2x4 WMMA tiles), K-step 32.
// AMODE: 0 = f32 A (convert), 1 = f16 A (raw copy)
// OMODE: 0 = f16 out [M,N], 1 = f16 out transposed per head [B][H][HD][S],
//        2 = f32 out [M,N]
// ---------------------------------------------------------------------------
template <int AMODE, int OMODE>
__global__ __launch_bounds__(256) void gemm_bias(
    const void* __restrict__ Ain, const float* __restrict__ W,
    const float* __restrict__ bias, void* __restrict__ Cout)
{
    __shared__ __align__(16) _Float16 As[128][40]; // [m][k], +8 pad
    __shared__ __align__(16) _Float16 Ws[128][40]; // transposed [n][k]

    const int tid   = threadIdx.x;
    const int lane  = tid & 31;
    const int wid   = tid >> 5;
    const int wm    = wid & 3;      // 4 waves in M  -> +32 rows
    const int wn    = wid >> 2;     // 2 waves in N  -> +64 cols
    const int l15   = lane & 15;
    const int hiH   = lane >> 4;
    const int m_blk = blockIdx.y * 128;
    const int n_blk = blockIdx.x * 128;

    v8f acc[2][4];
#pragma unroll
    for (int i = 0; i < 2; i++)
#pragma unroll
        for (int j = 0; j < 4; j++) acc[i][j] = (v8f)0.0f;

    for (int k0 = 0; k0 < GK; k0 += 32) {
        // ---- stage A tile 128x32 ----
        if (AMODE == 0) {
            const float* A = (const float*)Ain;
            for (int idx = tid; idx < 128 * 8; idx += 256) {
                const int row = idx >> 3;
                const int c4  = idx & 7;
                const float4 v = *(const float4*)(A + (size_t)(m_blk + row) * GK + k0 + c4 * 4);
                _Float16* dst = &As[row][c4 * 4];
                dst[0] = (_Float16)v.x; dst[1] = (_Float16)v.y;
                dst[2] = (_Float16)v.z; dst[3] = (_Float16)v.w;
            }
        } else {
            const _Float16* A = (const _Float16*)Ain;
            for (int idx = tid; idx < 128 * 4; idx += 256) {
                const int row = idx >> 2;
                const int c8  = idx & 3;
                *(half8*)&As[row][c8 * 8] =
                    *(const half8*)(A + (size_t)(m_blk + row) * GK + k0 + c8 * 8);
            }
        }
        // ---- stage W tile 32x128 transposed -> Ws[n][k] ----
        for (int idx = tid; idx < 32 * 32; idx += 256) {
            const int kk = idx >> 5;
            const int c4 = idx & 31;
            const float4 v = *(const float4*)(W + (size_t)(k0 + kk) * GN + n_blk + c4 * 4);
            Ws[c4 * 4 + 0][kk] = (_Float16)v.x;
            Ws[c4 * 4 + 1][kk] = (_Float16)v.y;
            Ws[c4 * 4 + 2][kk] = (_Float16)v.z;
            Ws[c4 * 4 + 3][kk] = (_Float16)v.w;
        }
        __syncthreads();

        const int kb = hiH * 8;   // A-frag: runs [kb..kb+7], [kb+16..kb+23]
        const int kg = hiH * 8;   // B-frag: runs [kg..kg+7], [kg+16..kg+23]
        v16h af[2], bf[4];
#pragma unroll
        for (int mi = 0; mi < 2; mi++) {
            const int row = wm * 32 + mi * 16 + l15;
            half8 lo = *(const half8*)&As[row][kb];
            half8 hi = *(const half8*)&As[row][16 + kb];
            af[mi] = CAT16(lo, hi);
        }
#pragma unroll
        for (int ni = 0; ni < 4; ni++) {
            const int col = wn * 64 + ni * 16 + l15;
            half8 lo = *(const half8*)&Ws[col][kg];
            half8 hi = *(const half8*)&Ws[col][kg + 16];
            bf[ni] = CAT16(lo, hi);
        }
#pragma unroll
        for (int mi = 0; mi < 2; mi++)
#pragma unroll
            for (int ni = 0; ni < 4; ni++)
                acc[mi][ni] = WMMA_F16(af[mi], bf[ni], acc[mi][ni]);
        __syncthreads();
    }

    // ---- epilogue: element (m = r + 8*half, n = l15) + bias ----
#pragma unroll
    for (int mi = 0; mi < 2; mi++)
#pragma unroll
        for (int ni = 0; ni < 4; ni++) {
            const int ng = n_blk + wn * 64 + ni * 16 + l15;
            const float bv = bias[ng];
#pragma unroll
            for (int r = 0; r < 8; r++) {
                const int mg = m_blk + wm * 32 + mi * 16 + r + 8 * hiH;
                const float val = acc[mi][ni][r] + bv;
                if (OMODE == 0) {
                    ((_Float16*)Cout)[(size_t)mg * GN + ng] = (_Float16)val;
                } else if (OMODE == 1) {
                    // V transposed per head: [b][h][hd][s]
                    const int b = mg >> 11, s = mg & (S_ - 1);
                    const int h = ng >> 6,  hd = ng & 63;
                    ((_Float16*)Cout)[(((size_t)(b * H_ + h) * 64 + hd) << 11) + s] =
                        (_Float16)val;
                } else {
                    ((float*)Cout)[(size_t)mg * GN + ng] = val;
                }
            }
        }
}

// ---------------------------------------------------------------------------
// Flash attention (causal): block = 128 threads = 4 waves, each wave owns a
// 16-row query strip. K/V tiles (64 keys) staged via TDM into padded LDS.
// ---------------------------------------------------------------------------
__global__ __launch_bounds__(128) void attn_flash_f16(
    const _Float16* __restrict__ Qp, const _Float16* __restrict__ Kp,
    const _Float16* __restrict__ Vt, _Float16* __restrict__ Ob)
{
    __shared__ __align__(16) _Float16 Ks[64][72];    // [key][hd]   (TDM, padded)
    __shared__ __align__(16) _Float16 Vs[64][72];    // [hd][key]   (TDM, padded)
    __shared__ __align__(16) _Float16 Ps[4][16][72]; // per-wave P strip [m][key]

    const int tid  = threadIdx.x;
    const int lane = tid & 31;
    const int w    = tid >> 5;
    const int l15  = lane & 15;
    const int hiH  = lane >> 4;
    const int bh   = blockIdx.y;
    const int b    = bh >> 4;
    const int h    = bh & 15;
    const int q0   = blockIdx.x * 64;
    const int qrow = q0 + w * 16;

    const unsigned int ks_off = (unsigned int)(uintptr_t)&Ks[0][0];
    const unsigned int vs_off = (unsigned int)(uintptr_t)&Vs[0][0];

    // preload Q strip as two 16x32 A-fragments, scaled by 1/sqrt(HD)=2^-3
    v16h qf[2];
    {
        const int kb = hiH * 8;
        const _Float16* qr = Qp + ((size_t)(b * S_) + qrow + l15) * D_ + h * 64;
#pragma unroll
        for (int f = 0; f < 2; f++) {
            half8 lo = *(const half8*)(qr + f * 32 + kb);
            half8 hi = *(const half8*)(qr + f * 32 + 16 + kb);
            lo *= (_Float16)0.125f;
            hi *= (_Float16)0.125f;
            qf[f] = CAT16(lo, hi);
        }
    }

    v8f o[4];
#pragma unroll
    for (int i = 0; i < 4; i++) o[i] = (v8f)0.0f;
    float ml[8], ll[8];
#pragma unroll
    for (int r = 0; r < 8; r++) { ml[r] = -1e30f; ll[r] = 0.0f; }

    for (int j0 = 0; j0 <= q0; j0 += 64) {   // causal: skip fully-masked tiles
        // ---- TDM: K tile [64 keys][64 hd], V tile [64 hd][64 keys] ----
        if (tid < 32) {
            tdm_load_tile_f16(ks_off,
                Kp + ((size_t)(b * S_) + j0) * D_ + h * 64, 64, 64, D_);
            tdm_load_tile_f16(vs_off,
                Vt + ((size_t)(b * H_ + h) * 64 << 11) + j0, 64, 64, S_);
            __builtin_amdgcn_s_wait_tensorcnt(0);
        }
        __syncthreads();

        // ---- scores: S[16x64] = Q(16x64) @ K^T ----
        v8f sc[4];
        const int kg = hiH * 8;
#pragma unroll
        for (int ns = 0; ns < 4; ns++) {
            v8f c = (v8f)0.0f;
            const int col = ns * 16 + l15;
#pragma unroll
            for (int ks = 0; ks < 2; ks++) {
                half8 lo = *(const half8*)&Ks[col][ks * 32 + kg];
                half8 hb = *(const half8*)&Ks[col][ks * 32 + kg + 16];
                v16h bfrag = CAT16(lo, hb);
                c = WMMA_F16(qf[ks], bfrag, c);
            }
            sc[ns] = c;
        }

        // ---- causal mask + online softmax (rows live in 16-lane halves) ----
        float rmax[8], rsum[8], corr[8];
#pragma unroll
        for (int r = 0; r < 8; r++) rmax[r] = -1e30f;
#pragma unroll
        for (int ns = 0; ns < 4; ns++) {
            const int key = j0 + ns * 16 + l15;
#pragma unroll
            for (int r = 0; r < 8; r++) {
                const int qg = qrow + r + 8 * hiH;
                float s = sc[ns][r];
                if (key > qg) s = -1e30f;
                sc[ns][r] = s;
                rmax[r] = fmaxf(rmax[r], s);
            }
        }
#pragma unroll
        for (int r = 0; r < 8; r++)
#pragma unroll
            for (int m = 8; m >= 1; m >>= 1)
                rmax[r] = fmaxf(rmax[r], __shfl_xor(rmax[r], m, 32));
#pragma unroll
        for (int r = 0; r < 8; r++) {
            const float mn = fmaxf(ml[r], rmax[r]);
            corr[r] = __expf(ml[r] - mn);
            ml[r] = mn;
            rsum[r] = 0.0f;
        }
#pragma unroll
        for (int ns = 0; ns < 4; ns++) {
#pragma unroll
            for (int r = 0; r < 8; r++) {
                const float p = __expf(sc[ns][r] - ml[r]);
                rsum[r] += p;
                Ps[w][r + 8 * hiH][ns * 16 + l15] = (_Float16)p;
            }
        }
#pragma unroll
        for (int r = 0; r < 8; r++)
#pragma unroll
            for (int m = 8; m >= 1; m >>= 1)
                rsum[r] += __shfl_xor(rsum[r], m, 32);
#pragma unroll
        for (int r = 0; r < 8; r++) ll[r] = ll[r] * corr[r] + rsum[r];
#pragma unroll
        for (int hs = 0; hs < 4; hs++)
#pragma unroll
            for (int r = 0; r < 8; r++) o[hs][r] *= corr[r];

        // ---- O += P(16x64) @ V(64x64) ----
        const int kb = hiH * 8;
#pragma unroll
        for (int ks = 0; ks < 2; ks++) {
            half8 plo = *(const half8*)&Ps[w][l15][ks * 32 + kb];
            half8 phi = *(const half8*)&Ps[w][l15][ks * 32 + kb + 16];
            v16h pf = CAT16(plo, phi);
#pragma unroll
            for (int hs = 0; hs < 4; hs++) {
                const int col = hs * 16 + l15;
                half8 vlo = *(const half8*)&Vs[col][ks * 32 + kg];
                half8 vhb = *(const half8*)&Vs[col][ks * 32 + kg + 16];
                v16h vf = CAT16(vlo, vhb);
                o[hs] = WMMA_F16(pf, vf, o[hs]);
            }
        }
        __syncthreads();
    }

    // ---- normalize, write f16 attn output [B,S,D] ----
#pragma unroll
    for (int hs = 0; hs < 4; hs++) {
#pragma unroll
        for (int r = 0; r < 8; r++) {
            const int qg = qrow + r + 8 * hiH;
            const size_t off = ((size_t)(b * S_) + qg) * D_ + h * 64 + hs * 16 + l15;
            Ob[off] = (_Float16)(o[hs][r] / ll[r]);
        }
    }
}

// ---------------------------------------------------------------------------
extern "C" void kernel_launch(void* const* d_in, const int* in_sizes, int n_in,
                              void* d_out, int out_size, void* d_ws, size_t ws_size,
                              hipStream_t stream) {
    const float* q  = (const float*)d_in[0];
    const float* k  = (const float*)d_in[1];
    const float* v  = (const float*)d_in[2];
    // d_in[3] = dense causal mask, handled analytically
    const float* Wq = (const float*)d_in[4];
    const float* bq = (const float*)d_in[5];
    const float* Wk = (const float*)d_in[6];
    const float* bk = (const float*)d_in[7];
    const float* Wv = (const float*)d_in[8];
    const float* bv = (const float*)d_in[9];
    const float* Wo = (const float*)d_in[10];
    const float* bo = (const float*)d_in[11];
    float* out = (float*)d_out;

    const size_t elems = (size_t)B_ * S_ * D_;
    _Float16* Qp = (_Float16*)d_ws;
    _Float16* Kp = Qp + elems;
    _Float16* Vt = Kp + elems;   // transposed per head: [B][H][HD][S]
    _Float16* Ab = Vt + elems;

    dim3 gg(GN / 128, GM / 128);  // (8, 64)
    gemm_bias<0, 0><<<gg, 256, 0, stream>>>(q, Wq, bq, Qp);
    gemm_bias<0, 0><<<gg, 256, 0, stream>>>(k, Wk, bk, Kp);
    gemm_bias<0, 1><<<gg, 256, 0, stream>>>(v, Wv, bv, Vt);
    attn_flash_f16<<<dim3(S_ / 64, B_ * H_), 128, 0, stream>>>(Qp, Kp, Vt, Ab);
    gemm_bias<1, 2><<<gg, 256, 0, stream>>>(Ab, Wo, bo, out);
}
